// Model_29016799052101
// MI455X (gfx1250) — matrix operand order repaired and checked by reading, not compiled, not run on hardware
//
#include <hip/hip_runtime.h>
#include <hip/hip_bf16.h>
#include <math.h>

// ---------------------------------------------------------------------------
// Types for CDNA5 WMMA (wave32): D(16x16,f32) = A(16x32,f16) x B(32x16,f16) + C
// ---------------------------------------------------------------------------
typedef __attribute__((ext_vector_type(16))) _Float16 v16h;
typedef __attribute__((ext_vector_type(8)))  _Float16 v8h;
typedef __attribute__((ext_vector_type(8)))  float    v8f;

union FragU { v8h h8[2]; v16h v; };

static constexpr int   TT      = 4096;   // output timesteps
static constexpr int   TALLOC  = 4176;   // padded rows per activation buffer
static constexpr int   NB      = 8;      // batch
static constexpr float RR_SLOPE = 0.22916666666666666f; // (1/8 + 1/3)/2

// ---------------------------------------------------------------------------
// CDNA5 async global->LDS copy (guarded; falls back to ds_store path).
// Builtin signature (from toolchain diagnostic): param1 = v4i addrspace(1)*,
// param2 = v4i addrspace(3)*, then imm offset, imm cpol.
// ---------------------------------------------------------------------------
#if defined(__has_builtin)
#  if __has_builtin(__builtin_amdgcn_global_load_async_to_lds_b128) && \
      __has_builtin(__builtin_amdgcn_s_wait_asynccnt)
#    define ATHENA_ASYNC_LDS 1
#  endif
#endif

#ifdef ATHENA_ASYNC_LDS
typedef int athena_b128 __attribute__((vector_size(16)));
typedef __attribute__((address_space(1))) athena_b128* athena_gp;
typedef __attribute__((address_space(3))) athena_b128* athena_lp;
#endif

__device__ __forceinline__ void copy16_g2s(_Float16* dst, const _Float16* src) {
#ifdef ATHENA_ASYNC_LDS
    __builtin_amdgcn_global_load_async_to_lds_b128(
        (athena_gp)(src), (athena_lp)(dst), 0, 0);
#else
    *(v8h*)dst = *(const v8h*)src;
#endif
}
__device__ __forceinline__ void wait_staging() {
#ifdef ATHENA_ASYNC_LDS
    __builtin_amdgcn_s_wait_asynccnt(0);
#endif
    __syncthreads();
}

// A fragment (16 rows x 32 K) from row-major source with given K-pitch.
// Lane<16: row=lane, K {k0..k0+7, k0+16..k0+23} with k0=0; lane>=16: k0=8.
// Works for both global and LDS pointers (inlined so addrspace is inferred).
__device__ __forceinline__ v16h load_a_frag(const _Float16* base, int stride, int lane) {
    const int row = lane & 15;
    const int k0  = (lane & 16) ? 8 : 0;
    const _Float16* p = base + (size_t)row * stride + k0;
    FragU u;
    u.h8[0] = *(const v8h*)(p);
    u.h8[1] = *(const v8h*)(p + 16);
    return u.v;
}

// B fragment (32 K x 16 out-channels). Source stored as [O][K] row-major.
// Lane<16: col=lane, K {0..7, 16..23}; lane>=16: K {8..15, 24..31}.
__device__ __forceinline__ v16h load_b_frag(const _Float16* base, int stride, int lane) {
    const int col = lane & 15;
    const int k0  = (lane & 16) ? 8 : 0;
    const _Float16* p = base + (size_t)col * stride + k0;
    FragU u;
    u.h8[0] = *(const v8h*)(p);
    u.h8[1] = *(const v8h*)(p + 16);
    return u.v;
}

__device__ __forceinline__ v8f wmma16(v16h a, v16h b, v8f c) {
    return __builtin_amdgcn_wmma_f32_16x16x32_f16(
        /*neg_a=*/false, a, /*neg_b=*/false, b,
        /*c_mod=*/(short)0, c, /*reuse_a=*/false, /*reuse_b=*/false);
}

// ---------------------------------------------------------------------------
// K0: weight preprocessing (f32 -> f16, tap-split convs, flattened mixtures)
// ---------------------------------------------------------------------------
__global__ void prep_weights(const float* __restrict__ cw0, const float* __restrict__ cw1,
                             const float* __restrict__ cw2,
                             const float* __restrict__ pw0, const float* __restrict__ pb0,
                             const float* __restrict__ pw1, const float* __restrict__ pb1,
                             _Float16* w00, _Float16* w01, _Float16* w10, _Float16* w11,
                             _Float16* w20, _Float16* w21,
                             _Float16* w0f, _Float16* bw0, _Float16* w1f, _Float16* bw1) {
    const int tid = blockIdx.x * blockDim.x + threadIdx.x;
    const int nth = gridDim.x * blockDim.x;
    for (int n = tid; n < 128 * 32; n += nth) {       // conv0 taps [O][I]
        w00[n] = (_Float16)cw0[n * 2 + 0];
        w01[n] = (_Float16)cw0[n * 2 + 1];
    }
    for (int n = tid; n < 128 * 128; n += nth) {      // conv1/conv2 taps
        w10[n] = (_Float16)cw1[n * 2 + 0];
        w11[n] = (_Float16)cw1[n * 2 + 1];
        w20[n] = (_Float16)cw2[n * 2 + 0];
        w21[n] = (_Float16)cw2[n * 2 + 1];
    }
    for (int n = tid; n < 256 * 1024; n += nth) {     // w0f[o][r*128+i] = pw0[r][o][i]
        const int o = n >> 10, rem = n & 1023, r = rem >> 7, i = rem & 127;
        w0f[n] = (_Float16)pw0[((size_t)(r * 256 + o)) * 128 + i];
    }
    for (int n = tid; n < 256 * 32; n += nth) {       // bw0[o][r] (padded to 32)
        const int o = n >> 5, r = n & 31;
        bw0[n] = (r < 8) ? (_Float16)pb0[r * 256 + o] : (_Float16)0.0f;
    }
    for (int n = tid; n < 64 * 2048; n += nth) {      // w1f[o][r*256+j] = pw1[r][o][j]
        const int o = n >> 11, rem = n & 2047, r = rem >> 8, j = rem & 255;
        w1f[n] = (_Float16)pw1[((size_t)(r * 64 + o)) * 256 + j];
    }
    for (int n = tid; n < 64 * 32; n += nth) {        // bw1[o][r]
        const int o = n >> 5, r = n & 31;
        bw1[n] = (r < 8) ? (_Float16)pb1[r * 64 + o] : (_Float16)0.0f;
    }
}

// ---------------------------------------------------------------------------
// K1: pad x left by 15 and transpose (B,C,T) -> time-major f16 [B][TALLOC][32]
// ---------------------------------------------------------------------------
__global__ void pad_x(const float* __restrict__ x, _Float16* __restrict__ X0) {
    const int tid = blockIdx.x * blockDim.x + threadIdx.x;
    const int nth = gridDim.x * blockDim.x;
    const int total = NB * 4111 * 32;
    for (int n = tid; n < total; n += nth) {
        const int i  = n & 31;
        const int bt = n >> 5;
        const int t  = bt % 4111;
        const int b  = bt / 4111;
        const float v = (t < 15) ? 0.0f : x[((size_t)(b * 32 + i)) * TT + (t - 15)];
        X0[((size_t)b * TALLOC + t) * 32 + i] = (_Float16)v;
    }
}

// ---------------------------------------------------------------------------
// K2-4: dilated conv (k=2) as two accumulating WMMA GEMM passes + bias + rrelu
//   grid: (ceil(Tout/32), Cout/64, B), block 256 (8 waves, 2x4 wave tiles)
//   Input rows t0b..t0b+31+dil staged once in LDS (reused by 4 N-waves x 2 taps)
// ---------------------------------------------------------------------------
__global__ void conv_gemm(const _Float16* __restrict__ in, _Float16* __restrict__ out,
                          const _Float16* __restrict__ wt0, const _Float16* __restrict__ wt1,
                          const float* __restrict__ bias,
                          int Cin, int Cout, int Tout, int dil) {
    __shared__ _Float16 sh[36 * 136];                 // up to 36 rows x (128+8)

    const int lane = threadIdx.x & 31;
    const int wv   = threadIdx.x >> 5;
    const int t0b  = blockIdx.x * 32;
    const int n0   = blockIdx.y * 64 + (wv >> 1) * 16;
    const int b    = blockIdx.z;
    const int lpitch = Cin + 8;                       // LDS row pitch (bank rotate)
    const _Float16* inb = in + (size_t)b * TALLOC * Cin;

    // --- stage input tile: (32+dil) rows x Cin halves ---
    {
        const int cw  = Cin >> 3;                     // 16B chunks per row
        const int nch = (32 + dil) * cw;
        for (int c = threadIdx.x; c < nch; c += 256) {
            const int row = c / cw;
            const int seg = c - row * cw;
            copy16_g2s(&sh[row * lpitch + seg * 8],
                       inb + (size_t)(t0b + row) * Cin + seg * 8);
        }
    }
    wait_staging();

    const int nkb = Cin >> 5;
    v8f acc = {};
    for (int tap = 0; tap < 2; ++tap) {
        const _Float16* abase = &sh[((wv & 1) * 16 + tap * dil) * lpitch];
        const _Float16* wt    = tap ? wt1 : wt0;
        for (int kb = 0; kb < nkb; ++kb) {
            v16h a  = load_a_frag(abase + kb * 32, lpitch, lane);
            v16h bf = load_b_frag(wt + (size_t)n0 * Cin + kb * 32, Cin, lane);
            acc = wmma16(a, bf, acc);
        }
    }
    const int t0  = t0b + (wv & 1) * 16;
    const int col = n0 + (lane & 15);
    const int tb  = t0 + ((lane & 16) ? 8 : 0);
    const float bz = bias[col];
    _Float16* outb = out + (size_t)b * TALLOC * Cout;
#pragma unroll
    for (int j = 0; j < 8; ++j) {
        const int t = tb + j;
        if (t < Tout) {
            float v = acc[j] + bz;
            v = (v >= 0.0f) ? v : RR_SLOPE * v;
            outb[(size_t)t * Cout + col] = (_Float16)v;
        }
    }
}

// ---------------------------------------------------------------------------
// K5: routing head: conv3(d=8) + rrelu + log-softmax + gumbel softmax(T=0.1)
//   writes gen_z (f32, (B,R,T) region of d_out) and zero-padded z [B][T][32] f16
// ---------------------------------------------------------------------------
__global__ void route_kernel(const _Float16* __restrict__ pre,
                             const float* __restrict__ w3, const float* __restrict__ b3,
                             const float* __restrict__ u,
                             float* __restrict__ genz, _Float16* __restrict__ zpad) {
    const int idx = blockIdx.x * blockDim.x + threadIdx.x;
    if (idx >= NB * TT) return;
    const int b = idx >> 12;
    const int t = idx & (TT - 1);
    const _Float16* p0 = pre + ((size_t)b * TALLOC + t) * 128;
    const _Float16* p1 = p0 + 8 * 128;

    float hv[8];
#pragma unroll
    for (int r = 0; r < 8; ++r) hv[r] = b3[r];
    for (int i = 0; i < 128; ++i) {
        const float a0 = (float)p0[i];
        const float a1 = (float)p1[i];
#pragma unroll
        for (int r = 0; r < 8; ++r) {
            const int wi = (r * 128 + i) * 2;
            hv[r] += a0 * w3[wi] + a1 * w3[wi + 1];
        }
    }
    float m = -1e30f;
#pragma unroll
    for (int r = 0; r < 8; ++r) {
        hv[r] = (hv[r] >= 0.0f) ? hv[r] : RR_SLOPE * hv[r];
        m = fmaxf(m, hv[r]);
    }
    float s = 0.0f;
#pragma unroll
    for (int r = 0; r < 8; ++r) s += expf(hv[r] - m);
    const float lse = m + logf(s);

    float y[8];
    float m2 = -1e30f;
#pragma unroll
    for (int r = 0; r < 8; ++r) {
        const float g = -logf(-logf(u[(size_t)idx * 8 + r]));
        y[r] = (hv[r] - lse + g) * 10.0f;   // / GUMBEL_TEMP
        m2 = fmaxf(m2, y[r]);
    }
    float s2 = 0.0f;
#pragma unroll
    for (int r = 0; r < 8; ++r) { y[r] = expf(y[r] - m2); s2 += y[r]; }
    const float inv = 1.0f / s2;

    _Float16* zp = zpad + (size_t)idx * 32;
#pragma unroll
    for (int r = 0; r < 8; ++r) {
        const float zr = y[r] * inv;
        genz[((size_t)(b * 8 + r)) * TT + t] = zr;
        zp[r] = (_Float16)zr;
    }
#pragma unroll
    for (int r = 8; r < 32; ++r) zp[r] = (_Float16)0.0f;
}

// ---------------------------------------------------------------------------
// K6: mixture layer 1: P1[t][o] = sum_{r,i} z[t][r]*pre[t][i]*pw0[r][o][i] + bias
//   Flat GEMM: M=4096(time) x N=256 x K=1024 (+32 bias block).
//   pre tile (32x128) + z tile staged in LDS; pre reused 8x (once per expert r).
// ---------------------------------------------------------------------------
__global__ void mix_gemm1(const _Float16* __restrict__ pre, const _Float16* __restrict__ zpad,
                          const _Float16* __restrict__ w0f, const _Float16* __restrict__ bw0,
                          _Float16* __restrict__ p1) {
    __shared__ _Float16 sh_pre[32 * 136];
    __shared__ _Float16 sh_z[32 * 40];

    const int lane = threadIdx.x & 31;
    const int wv   = threadIdx.x >> 5;
    const int t0b  = blockIdx.x * 32;
    const int n0   = blockIdx.y * 64 + (wv >> 1) * 16;
    const int b    = blockIdx.z;
    const _Float16* preb = pre + ((size_t)b * TALLOC + 8) * 128;   // pre[:, :, 8:]
    const _Float16* zb   = zpad + (size_t)b * TT * 32;

    // --- stage pre tile (32 rows x 128) and z tile (32 rows x 32) ---
    for (int c = threadIdx.x; c < 32 * 16; c += 256) {
        const int row = c >> 4, seg = c & 15;
        copy16_g2s(&sh_pre[row * 136 + seg * 8],
                   preb + (size_t)(t0b + row) * 128 + seg * 8);
    }
    for (int c = threadIdx.x; c < 32 * 4; c += 256) {
        const int row = c >> 2, seg = c & 3;
        copy16_g2s(&sh_z[row * 40 + seg * 8],
                   zb + (size_t)(t0b + row) * 32 + seg * 8);
    }
    wait_staging();

    const int t0loc = (wv & 1) * 16;
    const _Float16* zrow = &sh_z[(t0loc + (lane & 15)) * 40];

    v8f acc = {};
#pragma unroll 4
    for (int kb = 0; kb < 32; ++kb) {
        const int r  = kb >> 2;
        const int i0 = (kb & 3) << 5;
        v16h a = load_a_frag(&sh_pre[t0loc * 136 + i0], 136, lane);
        const _Float16 zv = zrow[r];
#pragma unroll
        for (int q = 0; q < 16; ++q) a[q] *= zv;
        v16h bf = load_b_frag(w0f + (size_t)n0 * 1024 + kb * 32, 1024, lane);
        acc = wmma16(a, bf, acc);
    }
    { // mixed-bias K block: A = z tile rows, B = bw0
        v16h a  = load_a_frag(&sh_z[t0loc * 40], 40, lane);
        v16h bf = load_b_frag(bw0 + (size_t)n0 * 32, 32, lane);
        acc = wmma16(a, bf, acc);
    }
    const int col = n0 + (lane & 15);
    const int tb  = t0b + t0loc + ((lane & 16) ? 8 : 0);
    _Float16* p1b = p1 + (size_t)b * TT * 256;
#pragma unroll
    for (int j = 0; j < 8; ++j)
        p1b[(size_t)(tb + j) * 256 + col] = (_Float16)acc[j];
}

// ---------------------------------------------------------------------------
// K7: mixture layer 2 + final outputs.
//   M=4096 x N=64 x K=2048 (+32 bias). col<32 -> mus; col>=32 -> sigmas(tanh)
//   P1 tile (32x256) + z tile staged in LDS; P1 reused 8x (once per expert r).
// ---------------------------------------------------------------------------
__global__ void mix_gemm2(const _Float16* __restrict__ p1, const _Float16* __restrict__ zpad,
                          const _Float16* __restrict__ w1f, const _Float16* __restrict__ bw1,
                          float* __restrict__ outbuf) {
    __shared__ _Float16 sh_p1[32 * 264];
    __shared__ _Float16 sh_z[32 * 40];

    const int lane = threadIdx.x & 31;
    const int wv   = threadIdx.x >> 5;
    const int t0b  = blockIdx.x * 32;
    const int n0   = (wv >> 1) * 16;
    const int b    = blockIdx.z;
    const _Float16* p1b = p1 + (size_t)b * TT * 256;
    const _Float16* zb  = zpad + (size_t)b * TT * 32;

    for (int c = threadIdx.x; c < 32 * 32; c += 256) {
        const int row = c >> 5, seg = c & 31;
        copy16_g2s(&sh_p1[row * 264 + seg * 8],
                   p1b + (size_t)(t0b + row) * 256 + seg * 8);
    }
    for (int c = threadIdx.x; c < 32 * 4; c += 256) {
        const int row = c >> 2, seg = c & 3;
        copy16_g2s(&sh_z[row * 40 + seg * 8],
                   zb + (size_t)(t0b + row) * 32 + seg * 8);
    }
    wait_staging();

    const int t0loc = (wv & 1) * 16;
    const _Float16* zrow = &sh_z[(t0loc + (lane & 15)) * 40];

    v8f acc = {};
#pragma unroll 4
    for (int kb = 0; kb < 64; ++kb) {
        const int r  = kb >> 3;
        const int j0 = (kb & 7) << 5;
        v16h a = load_a_frag(&sh_p1[t0loc * 264 + j0], 264, lane);
        const _Float16 zv = zrow[r];
#pragma unroll
        for (int q = 0; q < 16; ++q) a[q] *= zv;
        v16h bf = load_b_frag(w1f + (size_t)n0 * 2048 + kb * 32, 2048, lane);
        acc = wmma16(a, bf, acc);
    }
    {
        v16h a  = load_a_frag(&sh_z[t0loc * 40], 40, lane);
        v16h bf = load_b_frag(bw1 + (size_t)n0 * 32, 32, lane);
        acc = wmma16(a, bf, acc);
    }
    const int col = n0 + (lane & 15);
    const int tb  = t0b + t0loc + ((lane & 16) ? 8 : 0);
    typedef __attribute__((ext_vector_type(8))) float f8;
    if (col < 32) {
        f8 v;
#pragma unroll
        for (int j = 0; j < 8; ++j) v[j] = acc[j];
        *(f8*)(outbuf + ((size_t)(b * 32 + col)) * TT + tb) = v;
    } else {
        f8 v;
#pragma unroll
        for (int j = 0; j < 8; ++j) v[j] = 0.5f + 0.5f * tanhf(acc[j]) + 0.001f;
        *(f8*)(outbuf + (size_t)NB * 32 * TT + ((size_t)(b * 32 + (col - 32))) * TT + tb) = v;
    }
}

// ---------------------------------------------------------------------------
// Host launcher
// ---------------------------------------------------------------------------
extern "C" void kernel_launch(void* const* d_in, const int* in_sizes, int n_in,
                              void* d_out, int out_size, void* d_ws, size_t ws_size,
                              hipStream_t stream) {
    const float* x   = (const float*)d_in[0];
    const float* u   = (const float*)d_in[1];
    const float* cw0 = (const float*)d_in[2];
    const float* cb0 = (const float*)d_in[3];
    const float* cw1 = (const float*)d_in[4];
    const float* cb1 = (const float*)d_in[5];
    const float* cw2 = (const float*)d_in[6];
    const float* cb2 = (const float*)d_in[7];
    const float* cw3 = (const float*)d_in[8];
    const float* cb3 = (const float*)d_in[9];
    const float* pw0 = (const float*)d_in[10];
    const float* pb0 = (const float*)d_in[11];
    const float* pw1 = (const float*)d_in[12];
    const float* pb1 = (const float*)d_in[13];
    float* out = (float*)d_out;

    char* ws = (char*)d_ws;
    size_t off = 0;
    auto alloc = [&](size_t bytes) -> char* {
        char* p = ws + off;
        off = (off + bytes + 255) & ~(size_t)255;
        return p;
    };
    _Float16* X0  = (_Float16*)alloc((size_t)NB * TALLOC * 32  * 2);
    _Float16* C0  = (_Float16*)alloc((size_t)NB * TALLOC * 128 * 2);
    _Float16* C1  = (_Float16*)alloc((size_t)NB * TALLOC * 128 * 2);
    _Float16* PRE = (_Float16*)alloc((size_t)NB * TALLOC * 128 * 2);
    _Float16* ZP  = (_Float16*)alloc((size_t)NB * TT * 32 * 2);
    _Float16* P1  = (_Float16*)alloc((size_t)NB * TT * 256 * 2);
    _Float16* w00 = (_Float16*)alloc(128 * 32 * 2);
    _Float16* w01 = (_Float16*)alloc(128 * 32 * 2);
    _Float16* w10 = (_Float16*)alloc(128 * 128 * 2);
    _Float16* w11 = (_Float16*)alloc(128 * 128 * 2);
    _Float16* w20 = (_Float16*)alloc(128 * 128 * 2);
    _Float16* w21 = (_Float16*)alloc(128 * 128 * 2);
    _Float16* w0f = (_Float16*)alloc(256 * 1024 * 2);
    _Float16* bw0 = (_Float16*)alloc(256 * 32 * 2);
    _Float16* w1f = (_Float16*)alloc(64 * 2048 * 2);
    _Float16* bw1 = (_Float16*)alloc(64 * 32 * 2);
    (void)ws_size; (void)in_sizes; (void)n_in; (void)out_size;

    prep_weights<<<128, 256, 0, stream>>>(cw0, cw1, cw2, pw0, pb0, pw1, pb1,
                                          w00, w01, w10, w11, w20, w21,
                                          w0f, bw0, w1f, bw1);

    {
        const int total = NB * 4111 * 32;
        pad_x<<<(total + 255) / 256, 256, 0, stream>>>(x, X0);
    }

    // conv stack: grid (ceil(T'/32), Cout/64, B)
    conv_gemm<<<dim3(129, 2, NB), 256, 0, stream>>>(X0, C0, w00, w01, cb0, 32, 128, 4110, 1);
    conv_gemm<<<dim3(129, 2, NB), 256, 0, stream>>>(C0, C1, w10, w11, cb1, 128, 128, 4108, 2);
    conv_gemm<<<dim3(129, 2, NB), 256, 0, stream>>>(C1, PRE, w20, w21, cb2, 128, 128, 4104, 4);

    // routing head + gumbel; gen_z lives at d_out + 2*B*32*T
    route_kernel<<<(NB * TT) / 256, 256, 0, stream>>>(PRE, cw3, cb3, u,
                                                      out + (size_t)2 * NB * 32 * TT, ZP);

    // mixture layers
    mix_gemm1<<<dim3(128, 4, NB), 256, 0, stream>>>(PRE, ZP, w0f, bw0, P1);
    mix_gemm2<<<dim3(128, 1, NB), 256, 0, stream>>>(P1, ZP, w1f, bw1, out);
}
